// GATConv_62036507623869
// MI455X (gfx1250) — matrix operand order repaired and checked by reading, not compiled, not run on hardware
//
#include <hip/hip_runtime.h>

typedef __attribute__((ext_vector_type(16))) _Float16 v16h;
typedef __attribute__((ext_vector_type(8)))  float    v8f;
typedef __attribute__((ext_vector_type(4)))  unsigned v4u;
typedef __attribute__((ext_vector_type(8)))  int      v8i;
typedef __attribute__((ext_vector_type(4)))  int      v4i;

#define FEATS 128   // IN == H*D == 128
#define HEADS 4
#define NEG_SLOPE 0.2f
#define AROW 132    // LDS row stride in dwords: 128 + 4 pad (TDM pad feature)

// ---------------------------------------------------------------------------
// Kernel 0: one-time pre-swizzle of W (f32 128x128) into B-fragment order
// (this code's contiguous k order, same as the A loads): index = ((tile*4 + kstep)*32 + lane)*16 + p
// where element p maps to W[kstep*32 + (lane/16)*16 + p][tile*16 + lane%16].
// Each lane then loads its fragment as one contiguous 32B run from L2.
// ---------------------------------------------------------------------------
__global__ __launch_bounds__(256)
void gat_wprep(const float* __restrict__ Wm, _Float16* __restrict__ wfrag) {
  int i = blockIdx.x * 256 + threadIdx.x;   // 0 .. 16383
  if (i >= 128 * 128) return;
  int p = i & 15;
  int L = (i >> 4) & 31;
  int s = (i >> 9) & 3;
  int t = i >> 11;
  int k = (s << 5) + ((L >> 4) << 4) + p;
  int c = (t << 4) + (L & 15);
  wfrag[i] = (_Float16)Wm[k * 128 + c];
}

// ---------------------------------------------------------------------------
// Kernel 1: ft = feat @ W.  256 threads = 8 wave32; wave w owns cols
// [16w, 16w+16) of the 16-node row tile.  A tile (16x128 f32) arrives via the
// Tensor Data Mover into LDS with 4-dword row padding; B fragments stream
// from the pre-swizzled global buffer (L2 resident).
// ---------------------------------------------------------------------------
__global__ __launch_bounds__(256)
void gat_proj_wmma(const float* __restrict__ feat,
                   const _Float16* __restrict__ wfrag,
                   float* __restrict__ ft, int n) {
  __shared__ float sA[16 * AROW];           // 8448 B
  const int tid  = threadIdx.x;
  const int wave = tid >> 5;
  const int lane = tid & 31;
  const int m0   = blockIdx.x << 4;

  if (wave == 0) {
    // Build D# (ISA 8.3/8.4): 2D tile, 4-byte elems, tile 128x16,
    // pad_enable, pad_interval=6 (128 DW), pad_amount=3 (4 DW) -> row stride 132 DW.
    unsigned lds_off = (unsigned)(unsigned long long)(void*)sA; // low 32b = LDS offset
    unsigned long long ga =
        (unsigned long long)(const void*)(feat + (size_t)m0 * FEATS);
    unsigned td1 = (unsigned)(n - m0);      // rows remaining from tile start
    v4u g0 = { 1u,                          // count=1, user descriptor
               lds_off,
               (unsigned)(ga & 0xFFFFFFFFull),
               (unsigned)((ga >> 32) & 0x01FFFFFFull) | (2u << 30) }; // type=2
    v8i g1 = { (int)((2u << 16) | (1u << 20) | (6u << 22) | (3u << 25)),
               (int)(128u << 16),                              // tensor_dim0 lo16
               (int)((td1 & 0xFFFFu) << 16),                   // tensor_dim1 lo16
               (int)(((td1 >> 16) & 0xFFFFu) | (128u << 16)),  // dim1 hi | tile_dim0
               16,                                             // tile_dim1
               128,                                            // tensor_dim0_stride
               0, 0 };
    v4i gz  = { 0, 0, 0, 0 };
    v8i gz8 = { 0, 0, 0, 0, 0, 0, 0, 0 };
    __builtin_amdgcn_tensor_load_to_lds(g0, g1, gz, gz, gz8, 0);
    __builtin_amdgcn_s_wait_tensorcnt((short)0);
  }
  __syncthreads();

  const int mrow = lane & 15;               // A: lane%16 selects matrix row M
  const int half = lane >> 4;               // lane half selects K subrange
  const int ncol = (wave << 4) + mrow;      // B/D: column
  const float* rowp = sA + mrow * AROW;
  v8f acc = {};
#pragma unroll
  for (int s = 0; s < 4; ++s) {             // K = 128 in 4 steps of 32
    const int kb = s << 5;
    const float* p0 = rowp + kb + (half << 4);
    const float4 r0 = *(const float4*)(p0);
    const float4 r1 = *(const float4*)(p0 + 4);
    const float4 r2 = *(const float4*)(p0 + 8);
    const float4 r3 = *(const float4*)(p0 + 12);
    v16h a;
    a[0]  = (_Float16)r0.x;  a[1]  = (_Float16)r0.y;
    a[2]  = (_Float16)r0.z;  a[3]  = (_Float16)r0.w;
    a[4]  = (_Float16)r1.x;  a[5]  = (_Float16)r1.y;
    a[6]  = (_Float16)r1.z;  a[7]  = (_Float16)r1.w;
    a[8]  = (_Float16)r2.x;  a[9]  = (_Float16)r2.y;
    a[10] = (_Float16)r2.z;  a[11] = (_Float16)r2.w;
    a[12] = (_Float16)r3.x;  a[13] = (_Float16)r3.y;
    a[14] = (_Float16)r3.z;  a[15] = (_Float16)r3.w;
    const v16h b =
        *(const v16h*)(wfrag + ((((wave << 2) + s) << 5) + lane) * 16);
    acc = __builtin_amdgcn_wmma_f32_16x16x32_f16(
        /*neg_a=*/false, a, /*neg_b=*/false, b,
        /*c_mod=*/(short)0, acc, /*reuse_a=*/false, /*reuse_b=*/false);
  }
  // D layout: VGPR r holds row M = r + 8*(lane/16), col = lane%16
#pragma unroll
  for (int r = 0; r < 8; ++r) {
    int m = m0 + r + (half << 3);
    if (m < n) ft[m * FEATS + ncol] = acc[r];
  }
}

// ---------------------------------------------------------------------------
// Kernel 2: el[n,h] = sum_d ft[n,h,d]*attn_l[h,d]; same for er.
// One wave per node; lane L covers d-index 4L..4L+3 (head = L/8),
// 8-lane group reduction via shfl_xor.
// ---------------------------------------------------------------------------
__global__ __launch_bounds__(256)
void gat_elr(const float* __restrict__ ft, const float* __restrict__ attn_l,
             const float* __restrict__ attn_r, float* __restrict__ el,
             float* __restrict__ er, int n) {
  int t = blockIdx.x * 256 + threadIdx.x;
  int node = t >> 5;
  int lane = t & 31;
  if (node >= n) return;
  const float4 f  = *(const float4*)(ft + (size_t)node * FEATS + lane * 4);
  const float4 al = *(const float4*)(attn_l + lane * 4);
  const float4 ar = *(const float4*)(attn_r + lane * 4);
  float sl = f.x * al.x + f.y * al.y + f.z * al.z + f.w * al.w;
  float sr = f.x * ar.x + f.y * ar.y + f.z * ar.z + f.w * ar.w;
#pragma unroll
  for (int off = 4; off >= 1; off >>= 1) {
    sl += __shfl_xor(sl, off, 32);
    sr += __shfl_xor(sr, off, 32);
  }
  if ((lane & 7) == 0) {
    el[node * HEADS + (lane >> 3)] = sl;
    er[node * HEADS + (lane >> 3)] = sr;
  }
}

// monotone float<->uint key so segment-max can use integer atomicMax
__device__ __forceinline__ unsigned enc_key(float f) {
  unsigned u = __float_as_uint(f);
  return (u & 0x80000000u) ? ~u : (u | 0x80000000u);
}
__device__ __forceinline__ float dec_key(unsigned key) {
  unsigned u = (key & 0x80000000u) ? (key ^ 0x80000000u) : ~key;
  return __uint_as_float(u);
}

// ---------------------------------------------------------------------------
// Kernel 3: per (edge,head): e = leakyrelu(el[src]+er[dst]); segment max.
// ---------------------------------------------------------------------------
__global__ __launch_bounds__(256)
void gat_edge_score(const float* __restrict__ el, const float* __restrict__ er,
                    const int* __restrict__ src, const int* __restrict__ dst,
                    float* __restrict__ ebuf, unsigned* __restrict__ maxkey,
                    int e) {
  int t = blockIdx.x * 256 + threadIdx.x;
  if (t >= e * HEADS) return;
  int edge = t >> 2, h = t & 3;
  int s = src[edge], d = dst[edge];
  float v = el[s * HEADS + h] + er[d * HEADS + h];
  v = (v > 0.0f) ? v : NEG_SLOPE * v;
  ebuf[t] = v;
  atomicMax(&maxkey[d * HEADS + h], enc_key(v));
}

// ---------------------------------------------------------------------------
// Kernel 4: ex = exp(e - max[dst]); segment sum of ex (in-place on ebuf).
// ---------------------------------------------------------------------------
__global__ __launch_bounds__(256)
void gat_edge_exp(const unsigned* __restrict__ maxkey,
                  const int* __restrict__ dst, float* __restrict__ ebuf,
                  float* __restrict__ den, int e) {
  int t = blockIdx.x * 256 + threadIdx.x;
  if (t >= e * HEADS) return;
  int edge = t >> 2, h = t & 3;
  int d = dst[edge];
  float m = dec_key(maxkey[d * HEADS + h]);
  float xv = __expf(ebuf[t] - m);
  ebuf[t] = xv;
  atomicAdd(&den[d * HEADS + h], xv);
}

// ---------------------------------------------------------------------------
// Kernel 5: out[dst] += ft[src] * (ex/den[dst]).  One wave per edge; lane L
// handles the float4 at d-index 4L (head = L/8).  ft+out live in the 192MB L2.
// ---------------------------------------------------------------------------
__global__ __launch_bounds__(256)
void gat_aggregate(const float* __restrict__ ft, const float* __restrict__ ex,
                   const float* __restrict__ den, const int* __restrict__ src,
                   const int* __restrict__ dst, float* __restrict__ out, int e) {
  int t = blockIdx.x * 256 + threadIdx.x;
  int edge = t >> 5;
  if (edge >= e) return;
  int lane = t & 31;
  int s = src[edge], d = dst[edge];
  int h = lane >> 3;
  float dn = den[d * HEADS + h];
  float a = ex[edge * HEADS + h] / ((dn == 0.0f) ? 1.0f : dn);
  const float4 f = *(const float4*)(ft + (size_t)s * FEATS + lane * 4);
  float* o = out + (size_t)d * FEATS + lane * 4;
  atomicAdd(o + 0, f.x * a);
  atomicAdd(o + 1, f.y * a);
  atomicAdd(o + 2, f.z * a);
  atomicAdd(o + 3, f.w * a);
}

extern "C" void kernel_launch(void* const* d_in, const int* in_sizes, int n_in,
                              void* d_out, int out_size, void* d_ws,
                              size_t ws_size, hipStream_t stream) {
  const float* feat   = (const float*)d_in[0];
  const float* Wm     = (const float*)d_in[1];
  const float* attn_l = (const float*)d_in[2];
  const float* attn_r = (const float*)d_in[3];
  const int*   src    = (const int*)d_in[4];
  const int*   dst    = (const int*)d_in[5];
  const int n = in_sizes[0] / FEATS;   // 100000
  const int e = in_sizes[4];           // 1600000
  float* out = (float*)d_out;

  // workspace layout (256B aligned): wfrag | ft | el | er | ebuf | maxkey | den
  char* ws = (char*)d_ws;
  size_t off = 0;
  auto take = [&](size_t bytes) -> void* {
    void* p = ws + off;
    off = (off + bytes + 255) & ~(size_t)255;
    return p;
  };
  _Float16* wfrag  = (_Float16*)take((size_t)128 * 128 * sizeof(_Float16));
  float*    ft     = (float*)take((size_t)n * FEATS * sizeof(float));
  float*    el     = (float*)take((size_t)n * HEADS * sizeof(float));
  float*    er     = (float*)take((size_t)n * HEADS * sizeof(float));
  float*    ebuf   = (float*)take((size_t)e * HEADS * sizeof(float));
  unsigned* maxkey = (unsigned*)take((size_t)n * HEADS * sizeof(unsigned));
  float*    den    = (float*)take((size_t)n * HEADS * sizeof(float));

  (void)hipMemsetAsync(out, 0, (size_t)out_size * sizeof(float), stream);
  (void)hipMemsetAsync(maxkey, 0, (size_t)n * HEADS * sizeof(unsigned), stream);
  (void)hipMemsetAsync(den, 0, (size_t)n * HEADS * sizeof(float), stream);

  gat_wprep<<<dim3(64), dim3(256), 0, stream>>>(Wm, wfrag);
  gat_proj_wmma<<<dim3((n + 15) / 16), dim3(256), 0, stream>>>(feat, wfrag, ft, n);
  gat_elr<<<dim3((n * 32 + 255) / 256), dim3(256), 0, stream>>>(ft, attn_l,
                                                                attn_r, el, er, n);
  gat_edge_score<<<dim3((e * HEADS + 255) / 256), dim3(256), 0, stream>>>(
      el, er, src, dst, ebuf, maxkey, e);
  gat_edge_exp<<<dim3((e * HEADS + 255) / 256), dim3(256), 0, stream>>>(
      maxkey, dst, ebuf, den, e);
  gat_aggregate<<<dim3((int)(((long long)e * 32 + 255) / 256)), dim3(256), 0,
                  stream>>>(ft, ebuf, den, src, dst, out, e);
}
